// MemoryMoE2_73967926772421
// MI455X (gfx1250) — matrix operand order repaired and checked by reading, not compiled, not run on hardware
//
#include <hip/hip_runtime.h>
#include <math.h>

// ---------------- problem constants ----------------
#define N_SEQ   16384
#define DIM     1024
#define HID     2048
#define NEXP    16
#define TOPK    2

#define BLOCK_M 16                 // tokens per block tile
#define SLAB    (N_SEQ * TOPK)     // worst-case slots per expert
#define MAX_TILES (SLAB / BLOCK_M) // 2048
#define CHUNK_H 256                // H processed per fused chunk
#define NCHUNK  (HID / CHUNK_H)    // 8

typedef __attribute__((ext_vector_type(16))) __bf16 v16bf;
typedef __attribute__((ext_vector_type(8)))  float  v8f;

union V16U {
  v16bf v;
  unsigned short s[16];
  uint4 q[2];
};

// fp32 -> bf16 round-to-nearest-even
__device__ __forceinline__ unsigned short f2bf(float f) {
  union { float f; unsigned int u; } t; t.f = f;
  unsigned int u = t.u;
  u += 0x7FFFu + ((u >> 16) & 1u);
  return (unsigned short)(u >> 16);
}

// Branch-free tanh-form GELU (error ~1e-3, below bf16 WMMA rounding).
// Uses native v_exp_f32 / v_rcp_f32 — no EXEC divergence near WMMA.
__device__ __forceinline__ float gelu_fast(float v) {
  float u = 0.7978845608028654f * v * (1.0f + 0.044715f * v * v);
  u = fminf(fmaxf(u, -9.0f), 9.0f);
  float t = __builtin_amdgcn_exp2f(u * 2.8853900817779268f);   // e^{2u}
  float th = (t - 1.0f) * __builtin_amdgcn_rcpf(t + 1.0f);     // tanh(u)
  return 0.5f * v * (1.0f + th);
}

// 16-bit A-matrix (16x32) fragment slot for element (m, kk).
// lanes 0-15 hold K 0..15; lanes 16-31 hold K 16..31
__device__ __forceinline__ void a_slot(int m, int kk, int& lane, int& elem) {
  lane = m + ((kk >> 4) << 4);
  elem = kk & 15;
}

// Load one A fragment (16x32 bf16) from fragment-major LDS staging.
__device__ __forceinline__ v16bf load_a_frag(const unsigned short* lds, int kb) {
  int lane = threadIdx.x & 31;
  const uint4* p = (const uint4*)(lds + (((kb << 5) + lane) << 4));
  V16U r;
  r.q[0] = p[0];
  r.q[1] = p[1];
  return r.v;
}

// Slow path: B fragment (32x16) from row-major fp32 W with on-the-fly bf16 cvt.
__device__ __forceinline__ v16bf load_b_frag_f32(const float* __restrict__ W,
                                                 int k0, int n0, int ld) {
  int lane = threadIdx.x & 31;
  int n = n0 + (lane & 15);
  int kb = k0 + ((lane >> 4) << 4);
  const float* base = W + (size_t)kb * ld + n;
  V16U r;
#pragma unroll
  for (int e = 0; e < 16; ++e)
    r.s[e] = f2bf(base[(size_t)e * ld]);
  return r.v;
}

// Fast path: B fragment from pre-converted fragment-major bf16 weights.
// Layout: [nb][kb][lane][16 elems] per expert -> 2x global_load_b128 per lane.
__device__ __forceinline__ v16bf load_b_frag_pre(const unsigned short* __restrict__ Wf,
                                                 int kb, int nb, int KB) {
  int lane = threadIdx.x & 31;
  const uint4* p = (const uint4*)(Wf + (((size_t)(nb * KB + kb) * 32 + lane) << 4));
  V16U r;
  r.q[0] = p[0];
  r.q[1] = p[1];
  return r.v;
}

__device__ __forceinline__ v8f wmma_bf16(v16bf a, v16bf b, v8f c) {
  return __builtin_amdgcn_wmma_f32_16x16x32_bf16(
      false, a, false, b, (short)0, c, false, false);
}

// ---------------- kernel 1: zero output + counters ----------------
__global__ void MemoryMoE2_zero_kernel(float* __restrict__ y, int* __restrict__ counters,
                                       int n) {
  int i = blockIdx.x * blockDim.x + threadIdx.x;
  if (i < n) y[i] = 0.0f;
  if (i < 64) counters[i] = 0;
}

// ---------------- kernel 2: slot routing ----------------
__global__ void MemoryMoE2_route_kernel(const long long* __restrict__ ridx,
                                        const float* __restrict__ rw,
                                        int* __restrict__ counters,
                                        int* __restrict__ ptok,
                                        float* __restrict__ pgate) {
  int n = blockIdx.x * blockDim.x + threadIdx.x;
  if (n >= N_SEQ) return;
#pragma unroll
  for (int k = 0; k < TOPK; ++k) {
    int e = (int)ridx[(size_t)n * TOPK + k];
    float g = rw[(size_t)n * NEXP + e];
    int pos = atomicAdd(&counters[e], 1);
    ptok[(size_t)e * SLAB + pos]  = n;
    pgate[(size_t)e * SLAB + pos] = g;
  }
}

// ---------------- kernel 2b: convert weights to bf16 fragment-major ----------
// W: [NEXP][K][N] fp32, row-major. Wf: per expert, [nb][kb][lane][16] bf16.
// K*N == 1<<21 for both W1 (1024x2048) and W2 (2048x1024).
__global__ void MemoryMoE2_convw_kernel(const float* __restrict__ W,
                                        unsigned short* __restrict__ Wf,
                                        int logN) {
  const size_t total = (size_t)NEXP << 21;
  const int N = 1 << logN;
  const int KB = (1 << (21 - logN)) >> 5;     // K/32
  for (size_t i = (size_t)blockIdx.x * blockDim.x + threadIdx.x; i < total;
       i += (size_t)gridDim.x * blockDim.x) {
    int e = (int)(i >> 21);
    int rem = (int)(i & ((1u << 21) - 1u));
    int k = rem >> logN;
    int n = rem & (N - 1);
    int kk = k & 31, kb = k >> 5, nb = n >> 4;
    int lane = (n & 15) + ((kk >> 4) << 4);
    int elem = kk & 15;
    Wf[((size_t)e << 21) + (((size_t)(nb * KB + kb) * 32 + lane) << 4) + elem] =
        f2bf(W[i]);
  }
}

// ---------------- kernel 3: fused expert FFN ----------------
template <bool PRECONV>
__global__ __launch_bounds__(256)
void MemoryMoE2_moe_kernel(const float* __restrict__ x,
                           const float* __restrict__ W1,
                           const float* __restrict__ W2,
                           const unsigned short* __restrict__ W1f,
                           const unsigned short* __restrict__ W2f,
                           const int* __restrict__ counters,
                           const int* __restrict__ ptok,
                           const float* __restrict__ pgate,
                           float* __restrict__ y) {
  __shared__ unsigned short xA[32 * 32 * 16];  // 32 KB: x tile, A-fragment major
  __shared__ unsigned short hA[8 * 32 * 16];   //  8 KB: h chunk, A-fragment major
  __shared__ int   sTok[BLOCK_M];
  __shared__ float sGate[BLOCK_M];

  const int e = blockIdx.x / MAX_TILES;
  const int t = blockIdx.x % MAX_TILES;
  const int cnt = counters[e];
  if (t * BLOCK_M >= cnt) return;            // uniform early exit

  const float* __restrict__ W1e = W1 + ((size_t)e << 21);
  const float* __restrict__ W2e = W2 + ((size_t)e << 21);
  const unsigned short* __restrict__ W1fe = W1f + ((size_t)e << 21);
  const unsigned short* __restrict__ W2fe = W2f + ((size_t)e << 21);

  const int tid  = threadIdx.x;
  const int wave = tid >> 5;
  const int lane = tid & 31;

  // ---- load tile token list + gates ----
  if (tid < BLOCK_M) {
    int i = t * BLOCK_M + tid;
    if (i < cnt) {
      sTok[tid]  = ptok[(size_t)e * SLAB + i];
      sGate[tid] = pgate[(size_t)e * SLAB + i];
    } else {
      sTok[tid]  = -1;
      sGate[tid] = 0.0f;
    }
  }
  __syncthreads();

  // ---- stage x tile into LDS as bf16 A-fragments ----
  for (int i = tid; i < BLOCK_M * DIM; i += 256) {
    int m = i >> 10;            // DIM = 1024
    int k = i & (DIM - 1);
    int tok = sTok[m];
    float v = (tok >= 0) ? x[(size_t)tok * DIM + k] : 0.0f;
    int l2, e2;
    a_slot(m, k & 31, l2, e2);
    xA[(((k >> 5) << 5) + l2) * 16 + e2] = f2bf(v);
  }

  // persistent y accumulators: wave owns cols [wave*128, wave*128+128)
  v8f yacc[8];
#pragma unroll
  for (int nf = 0; nf < 8; ++nf) yacc[nf] = (v8f){};

  const int hiM = (lane >> 4) << 3;   // +8 row offset for lanes 16-31 (C layout)

#pragma unroll 1
  for (int c = 0; c < NCHUNK; ++c) {
    __syncthreads();   // xA ready (c==0); hA free for overwrite (c>0)

    // ---- stage 1: h[:, c*256 + wave*32 .. +32) = gelu(x @ W1) ----
    v8f acc0 = (v8f){}, acc1 = (v8f){};
    const int n0  = c * CHUNK_H + wave * 32;
    const int nb0 = n0 >> 4;
#pragma unroll 1
    for (int kb = 0; kb < DIM / 32; ++kb) {
      v16bf a, b0, b1;
      a = load_a_frag(xA, kb);
      if (PRECONV) {
        b0 = load_b_frag_pre(W1fe, kb, nb0,     DIM / 32);
        b1 = load_b_frag_pre(W1fe, kb, nb0 + 1, DIM / 32);
        // prefetch the nb0 fragment two K-blocks ahead (contiguous in this layout)
        __builtin_prefetch(W1fe + (((size_t)(nb0 * (DIM / 32) + kb + 2) * 32 + lane) << 4), 0, 1);
      } else {
        b0 = load_b_frag_f32(W1e, kb * 32, n0,      HID);
        b1 = load_b_frag_f32(W1e, kb * 32, n0 + 16, HID);
      }
      acc0 = wmma_bf16(a, b0, acc0);
      acc1 = wmma_bf16(a, b1, acc1);
    }

    // gelu + scatter into hA (chunk-local A-fragment layout)
#pragma unroll
    for (int r = 0; r < 8; ++r) {
      int m = r + hiM;
      {
        int kk2 = wave * 32 + (lane & 15);           // acc0 columns
        int l2, e2; a_slot(m, kk2 & 31, l2, e2);
        hA[((((kk2 >> 5) << 5)) + l2) * 16 + e2] = f2bf(gelu_fast(acc0[r]));
      }
      {
        int kk2 = wave * 32 + 16 + (lane & 15);      // acc1 columns
        int l2, e2; a_slot(m, kk2 & 31, l2, e2);
        hA[((((kk2 >> 5) << 5)) + l2) * 16 + e2] = f2bf(gelu_fast(acc1[r]));
      }
    }
    __syncthreads();   // hA ready for all waves

    // ---- stage 2: y[:, wave*128 ..] += h_chunk @ W2[c*256 .., :] ----
#pragma unroll 1
    for (int kb = 0; kb < CHUNK_H / 32; ++kb) {
      v16bf a = load_a_frag(hA, kb);
      int krow = c * CHUNK_H + kb * 32;   // global K row in W2
      int kbg  = krow >> 5;               // global kb in W2 fragment layout
#pragma unroll
      for (int nf = 0; nf < 8; ++nf) {
        v16bf b;
        if (PRECONV) {
          b = load_b_frag_pre(W2fe, kbg, wave * 8 + nf, HID / 32);
          __builtin_prefetch(W2fe + (((size_t)((wave * 8 + nf) * (HID / 32) + kbg + 1) * 32 + lane) << 4), 0, 1);
        } else {
          b = load_b_frag_f32(W2e, krow, wave * 128 + nf * 16, DIM);
        }
        yacc[nf] = wmma_bf16(a, b, yacc[nf]);
      }
    }
  }

  // ---- epilogue: gate-scale and accumulate into y ----
#pragma unroll
  for (int nf = 0; nf < 8; ++nf) {
#pragma unroll
    for (int r = 0; r < 8; ++r) {
      int m = r + hiM;
      int tok = sTok[m];
      float g = sGate[m];
      if (tok >= 0 && g != 0.0f) {
        int col = wave * 128 + nf * 16 + (lane & 15);
        atomicAdd(&y[(size_t)tok * DIM + col], g * yacc[nf][r]);
      }
    }
  }
}

// ---------------- launcher ----------------
extern "C" void kernel_launch(void* const* d_in, const int* in_sizes, int n_in,
                              void* d_out, int out_size, void* d_ws, size_t ws_size,
                              hipStream_t stream) {
  const float*     x    = (const float*)d_in[0];
  const float*     rw   = (const float*)d_in[1];
  const long long* ridx = (const long long*)d_in[2];
  const float*     W1   = (const float*)d_in[3];
  const float*     W2   = (const float*)d_in[4];
  float* y = (float*)d_out;

  // workspace layout:
  // [64 int counters][NEXP*SLAB int tokens][NEXP*SLAB float gates]
  // [NEXP*2^21 ushort W1f][NEXP*2^21 ushort W2f]
  char* ws = (char*)d_ws;
  int*            counters = (int*)ws;
  int*            ptok     = (int*)(ws + 256);
  float*          pgate    = (float*)(ws + 256 + (size_t)NEXP * SLAB * 4);
  size_t off_w1f = 256 + (size_t)NEXP * SLAB * 8;
  unsigned short* W1f      = (unsigned short*)(ws + off_w1f);
  unsigned short* W2f      = (unsigned short*)(ws + off_w1f + ((size_t)NEXP << 22));
  const size_t need = off_w1f + ((size_t)NEXP << 23);
  const bool preconv = ws_size >= need;

  const int n_out = N_SEQ * DIM;
  MemoryMoE2_zero_kernel<<<(n_out + 255) / 256, 256, 0, stream>>>(y, counters, n_out);
  MemoryMoE2_route_kernel<<<(N_SEQ + 255) / 256, 256, 0, stream>>>(ridx, rw, counters,
                                                                   ptok, pgate);
  if (preconv) {
    MemoryMoE2_convw_kernel<<<16384, 256, 0, stream>>>(W1, W1f, 11);  // N=HID=2048
    MemoryMoE2_convw_kernel<<<16384, 256, 0, stream>>>(W2, W2f, 10);  // N=DIM=1024
    MemoryMoE2_moe_kernel<true><<<NEXP * MAX_TILES, 256, 0, stream>>>(
        x, W1, W2, W1f, W2f, counters, ptok, pgate, y);
  } else {
    MemoryMoE2_moe_kernel<false><<<NEXP * MAX_TILES, 256, 0, stream>>>(
        x, W1, W2, W1f, W2f, counters, ptok, pgate, y);
  }
}
